// Attention_4363686773373
// MI455X (gfx1250) — matrix operand order repaired and checked by reading, not compiled, not run on hardware
//
#include <hip/hip_runtime.h>
#include <math.h>

// ---------------------------------------------------------------------------
// Problem constants: B=4, T=2048, D=768, H=12, HD=64
// ---------------------------------------------------------------------------
#define BT      8192        // B*T rows
#define DIM     768
#define NT      2048        // T
#define NH      12
#define HD      64
#define EPSN    1e-4f
#define GAIN_T  0.04066308f // 1.8402 / sqrt(2048)
#define INV_SQRT_D 0.036084392f // 1/sqrt(768)

typedef __bf16 bf16_t;
typedef __attribute__((ext_vector_type(16))) __bf16 v16bf;
typedef __attribute__((ext_vector_type(8)))  float  v8f;

__device__ __forceinline__ v8f wmma_bf16(v16bf a, v16bf b, v8f c) {
  // D = A(16x32 bf16) * B(32x16 bf16) + C(16x16 f32)
  return __builtin_amdgcn_wmma_f32_16x16x32_bf16(
      false, a, false, b, (short)0, c, false, false);
}

// A-fragment (16x32 bf16), row-major source with leading dim `ld`.
// lane L: row = m0 + (L&15); K elements {0..15} (+16 for upper half-wave).
__device__ __forceinline__ v16bf load_a_frag(const bf16_t* src, int ld, int m0, int k0) {
  const int lane = threadIdx.x & 31;
  const int row  = m0 + (lane & 15);
  const int koff = k0 + ((lane & 16) ? 16 : 0);
  const bf16_t* p = src + (size_t)row * ld + koff;
  v16bf a;
#pragma unroll
  for (int e = 0; e < 16; ++e) a[e] = p[e];
  return a;
}

// B-fragment (32x16 bf16). Source addressed as element(k,n) = src[n*ld + k]
// (i.e. K contiguous per column). lane L: col = n0 + (L&15); K {0..15}/{16..31}.
__device__ __forceinline__ v16bf load_b_colk(const bf16_t* src, int ld, int n0, int k0) {
  const int lane = threadIdx.x & 31;
  const int col  = n0 + (lane & 15);
  const int kb   = k0 + ((lane & 16) ? 16 : 0);
  const bf16_t* p = src + (size_t)col * ld + kb;
  v16bf b;
#pragma unroll
  for (int e = 0; e < 16; ++e) b[e] = p[e];
  return b;
}

// ---------------------------------------------------------------------------
// Kernel 1: normalize weight rows (w / (||row||+eps)), cast to bf16.
// blocks 0..2303 -> qkv_w rows, 2304..3071 -> out_w rows. 256 thr, 3 elem/thr.
// ---------------------------------------------------------------------------
__global__ __launch_bounds__(256)
void k_norm_weights(const float* __restrict__ qkv_w, const float* __restrict__ out_w,
                    bf16_t* __restrict__ wn, bf16_t* __restrict__ own) {
  const int row = blockIdx.x, tid = threadIdx.x;
  const float* src;
  bf16_t* dst;
  if (row < 3 * DIM) { src = qkv_w + (size_t)row * DIM;             dst = wn  + (size_t)row * DIM; }
  else               { src = out_w + (size_t)(row - 3 * DIM) * DIM; dst = own + (size_t)(row - 3 * DIM) * DIM; }
  float v[3], ss = 0.f;
#pragma unroll
  for (int i = 0; i < 3; ++i) { v[i] = src[tid + i * 256]; ss += v[i] * v[i]; }
  __shared__ float red[256];
  red[tid] = ss; __syncthreads();
#pragma unroll
  for (int off = 128; off > 0; off >>= 1) {
    if (tid < off) red[tid] += red[tid + off];
    __syncthreads();
  }
  const float inv = 1.f / (sqrtf(red[0]) + EPSN);
#pragma unroll
  for (int i = 0; i < 3; ++i) dst[tid + i * 256] = (bf16_t)(v[i] * inv);
}

// ---------------------------------------------------------------------------
// Kernel 2: cast x -> bf16 and compute per-token magnitude ||x||/sqrt(D).
// ---------------------------------------------------------------------------
__global__ __launch_bounds__(256)
void k_prep_x(const float* __restrict__ x, bf16_t* __restrict__ xb, float* __restrict__ mag) {
  const int row = blockIdx.x, tid = threadIdx.x;
  const float* src = x + (size_t)row * DIM;
  bf16_t* dst = xb + (size_t)row * DIM;
  float v[3], ss = 0.f;
#pragma unroll
  for (int i = 0; i < 3; ++i) { v[i] = src[tid + i * 256]; ss += v[i] * v[i]; }
  __shared__ float red[256];
  red[tid] = ss; __syncthreads();
#pragma unroll
  for (int off = 128; off > 0; off >>= 1) {
    if (tid < off) red[tid] += red[tid + off];
    __syncthreads();
  }
  if (tid == 0) mag[row] = sqrtf(red[0]) * INV_SQRT_D;
#pragma unroll
  for (int i = 0; i < 3; ++i) dst[tid + i * 256] = (bf16_t)v[i];
}

// ---------------------------------------------------------------------------
// Kernel 3: QKV GEMM (x @ wn^T). Block = 128(t) x 64(e = one head/kind) tile,
// 4 waves, each wave a 32x64 strip (2 A-frags, 8 accumulators, B reused x2).
// Fused cosine-normalize for q,k; attention scale folded into q.
// Output layout (B*H, T, 64) bf16.
// ---------------------------------------------------------------------------
__global__ __launch_bounds__(128)
void k_qkv_gemm(const bf16_t* __restrict__ xb, const bf16_t* __restrict__ wn,
                bf16_t* __restrict__ qo, bf16_t* __restrict__ ko, bf16_t* __restrict__ vo) {
  const int tid = threadIdx.x, lane = tid & 31, wv = tid >> 5;
  const int hh = (lane >> 4) & 1;          // half-wave
  const int tileT = blockIdx.x * 128;
  const int et = blockIdx.y;               // 0..35
  const int kidx = et / NH, head = et % NH;
  const int erow0 = kidx * DIM + head * HD;
  const int m0 = tileT + wv * 32;

  v8f acc[2][4] = {};
  for (int k0 = 0; k0 < DIM; k0 += 32) {
    v16bf a0 = load_a_frag(xb, DIM, m0,      k0);
    v16bf a1 = load_a_frag(xb, DIM, m0 + 16, k0);
#pragma unroll
    for (int j = 0; j < 4; ++j) {
      v16bf bw = load_b_colk(wn, DIM, erow0 + j * 16, k0);
      acc[0][j] = wmma_bf16(a0, bw, acc[0][j]);
      acc[1][j] = wmma_bf16(a1, bw, acc[1][j]);
    }
  }

  bf16_t* dst = (kidx == 0) ? qo : ((kidx == 1) ? ko : vo);
#pragma unroll
  for (int g = 0; g < 2; ++g) {
    float fs[8];
    if (kidx < 2) {
      float ss[8];
#pragma unroll
      for (int r = 0; r < 8; ++r)
        ss[r] = acc[g][0][r]*acc[g][0][r] + acc[g][1][r]*acc[g][1][r] +
                acc[g][2][r]*acc[g][2][r] + acc[g][3][r]*acc[g][3][r];
#pragma unroll
      for (int mask = 1; mask < 16; mask <<= 1)
#pragma unroll
        for (int r = 0; r < 8; ++r) ss[r] += __shfl_xor(ss[r], mask, 32);
      const float num = (kidx == 0) ? 1.0f : 8.0f;  // q: sqrt(HD)*scale=1, k: sqrt(HD)=8
#pragma unroll
      for (int r = 0; r < 8; ++r) fs[r] = num / (sqrtf(ss[r]) + EPSN);
    } else {
#pragma unroll
      for (int r = 0; r < 8; ++r) fs[r] = 1.f;
    }
#pragma unroll
    for (int r = 0; r < 8; ++r) {
      const int gm = m0 + g * 16 + r + hh * 8;       // global row in [0, B*T)
      const int b = gm >> 11, t = gm & (NT - 1);
      bf16_t* p = dst + ((size_t)(b * NH + head) * NT + t) * HD + (lane & 15);
#pragma unroll
      for (int j = 0; j < 4; ++j) p[j * 16] = (bf16_t)(acc[g][j][r] * fs[r]);
    }
  }
}

// ---------------------------------------------------------------------------
// Kernel 4: sigmoid attention, streaming over keys in 64-blocks.
// Block = 128 query rows of one (b,h); 4 waves, each a 32x64 output strip.
// Each K/V fragment feeds two WMMAs (row-group reuse). S staged in LDS (bf16)
// for A-fragment relayout; V staged transposed (stride 72 padding).
// Epilogue: *gain, per-row normalize, * per-token magnitude.
// ---------------------------------------------------------------------------
__global__ __launch_bounds__(128)
void k_attention(const bf16_t* __restrict__ q, const bf16_t* __restrict__ k,
                 const bf16_t* __restrict__ v, const float* __restrict__ mag,
                 bf16_t* __restrict__ att) {
  __shared__ bf16_t ldsS[128 * 72];
  __shared__ bf16_t ldsVt[64 * 72];
  const int tid = threadIdx.x, lane = tid & 31, wv = tid >> 5;
  const int hh = (lane >> 4) & 1;
  const int tileT = blockIdx.x * 128;
  const int bh = blockIdx.y;
  const int b = bh / NH, head = bh % NH;
  const bf16_t* qb = q + (size_t)bh * NT * HD;
  const bf16_t* kb = k + (size_t)bh * NT * HD;
  const bf16_t* vb = v + (size_t)bh * NT * HD;
  const int m0 = wv * 32;                  // wave strip within tile

  v16bf aQ[2][2];
#pragma unroll
  for (int g = 0; g < 2; ++g) {
    aQ[g][0] = load_a_frag(qb, HD, tileT + m0 + g * 16, 0);
    aQ[g][1] = load_a_frag(qb, HD, tileT + m0 + g * 16, 32);
  }

  v8f o[2][4] = {};
  for (int s0 = 0; s0 < NT; s0 += 64) {
    __syncthreads();                        // protect LDS reuse
    // stage V^T (64 keys x 64 dim) into LDS, padded stride 72
    for (int i = tid; i < 64 * 64; i += 128) {
      const int s = i >> 6, e = i & 63;
      ldsVt[e * 72 + s] = vb[(size_t)(s0 + s) * HD + e];
    }
    // S strip: scores for this wave's 32 query rows x 64 keys
#pragma unroll
    for (int j = 0; j < 4; ++j) {
      v16bf bk0 = load_b_colk(kb, HD, s0 + j * 16, 0);
      v16bf bk1 = load_b_colk(kb, HD, s0 + j * 16, 32);
#pragma unroll
      for (int g = 0; g < 2; ++g) {
        v8f sa = {};
        sa = wmma_bf16(aQ[g][0], bk0, sa);
        sa = wmma_bf16(aQ[g][1], bk1, sa);
#pragma unroll
        for (int r = 0; r < 8; ++r) {
          const int m = m0 + g * 16 + r + hh * 8;
          const float sg = 1.f / (1.f + __expf(-sa[r]));  // sigmoid (scale pre-folded)
          ldsS[m * 72 + j * 16 + (lane & 15)] = (bf16_t)sg;
        }
      }
    }
    __syncthreads();
    v16bf aS[2][2];
#pragma unroll
    for (int g = 0; g < 2; ++g) {
      aS[g][0] = load_a_frag(ldsS, 72, m0 + g * 16, 0);
      aS[g][1] = load_a_frag(ldsS, 72, m0 + g * 16, 32);
    }
#pragma unroll
    for (int j = 0; j < 4; ++j) {
      v16bf bv0 = load_b_colk(ldsVt, 72, j * 16, 0);
      v16bf bv1 = load_b_colk(ldsVt, 72, j * 16, 32);
#pragma unroll
      for (int g = 0; g < 2; ++g) {
        o[g][j] = wmma_bf16(aS[g][0], bv0, o[g][j]);
        o[g][j] = wmma_bf16(aS[g][1], bv1, o[g][j]);
      }
    }
  }

  // epilogue: gain, then magnitude-preserving normalize * per-token mag
#pragma unroll
  for (int g = 0; g < 2; ++g) {
#pragma unroll
    for (int j = 0; j < 4; ++j)
#pragma unroll
      for (int r = 0; r < 8; ++r) o[g][j][r] *= GAIN_T;

    float ss[8];
#pragma unroll
    for (int r = 0; r < 8; ++r)
      ss[r] = o[g][0][r]*o[g][0][r] + o[g][1][r]*o[g][1][r] +
              o[g][2][r]*o[g][2][r] + o[g][3][r]*o[g][3][r];
#pragma unroll
    for (int mask = 1; mask < 16; mask <<= 1)
#pragma unroll
      for (int r = 0; r < 8; ++r) ss[r] += __shfl_xor(ss[r], mask, 32);

#pragma unroll
    for (int r = 0; r < 8; ++r) {
      const int t = tileT + m0 + g * 16 + r + hh * 8;
      const float fs = mag[b * NT + t] * 8.f / (sqrtf(ss[r]) + EPSN);
      bf16_t* p = att + ((size_t)(b * NT + t)) * DIM + head * HD + (lane & 15);
#pragma unroll
      for (int j = 0; j < 4; ++j) p[j * 16] = (bf16_t)(o[g][j][r] * fs);
    }
  }
}

// ---------------------------------------------------------------------------
// Kernel 5: output projection (att @ own^T), f32 out. Same tiling as kernel 3.
// ---------------------------------------------------------------------------
__global__ __launch_bounds__(128)
void k_outproj(const bf16_t* __restrict__ att, const bf16_t* __restrict__ own,
               float* __restrict__ out) {
  const int tid = threadIdx.x, lane = tid & 31, wv = tid >> 5;
  const int hh = (lane >> 4) & 1;
  const int m0 = blockIdx.x * 128 + wv * 32;
  const int e0 = blockIdx.y * 64;

  v8f acc[2][4] = {};
  for (int k0 = 0; k0 < DIM; k0 += 32) {
    v16bf a0 = load_a_frag(att, DIM, m0,      k0);
    v16bf a1 = load_a_frag(att, DIM, m0 + 16, k0);
#pragma unroll
    for (int j = 0; j < 4; ++j) {
      v16bf bw = load_b_colk(own, DIM, e0 + j * 16, k0);
      acc[0][j] = wmma_bf16(a0, bw, acc[0][j]);
      acc[1][j] = wmma_bf16(a1, bw, acc[1][j]);
    }
  }
#pragma unroll
  for (int g = 0; g < 2; ++g)
#pragma unroll
    for (int r = 0; r < 8; ++r) {
      float* p = out + (size_t)(m0 + g * 16 + r + hh * 8) * DIM + e0 + (lane & 15);
#pragma unroll
      for (int j = 0; j < 4; ++j) p[j * 16] = acc[g][j][r];
    }
}

// ---------------------------------------------------------------------------
extern "C" void kernel_launch(void* const* d_in, const int* in_sizes, int n_in,
                              void* d_out, int out_size, void* d_ws, size_t ws_size,
                              hipStream_t stream) {
  const float* x     = (const float*)d_in[0];
  const float* qkv_w = (const float*)d_in[1];
  const float* out_w = (const float*)d_in[2];
  float* out = (float*)d_out;

  char* ws = (char*)d_ws;
  size_t off = 0;
  auto wsalloc = [&](size_t bytes) -> void* {
    void* p = ws + off;
    off = (off + bytes + 255) & ~(size_t)255;
    return p;
  };
  bf16_t* xb  = (bf16_t*)wsalloc((size_t)BT * DIM * 2);
  bf16_t* wn  = (bf16_t*)wsalloc((size_t)3 * DIM * DIM * 2);
  bf16_t* own = (bf16_t*)wsalloc((size_t)DIM * DIM * 2);
  bf16_t* qb  = (bf16_t*)wsalloc((size_t)4 * NH * NT * HD * 2);
  bf16_t* kb  = (bf16_t*)wsalloc((size_t)4 * NH * NT * HD * 2);
  bf16_t* vb  = (bf16_t*)wsalloc((size_t)4 * NH * NT * HD * 2);
  bf16_t* att = (bf16_t*)wsalloc((size_t)BT * DIM * 2);
  float*  mg  = (float*)wsalloc((size_t)BT * 4);

  k_norm_weights<<<dim3(4 * DIM), 256, 0, stream>>>(qkv_w, out_w, wn, own);
  k_prep_x     <<<dim3(BT), 256, 0, stream>>>(x, xb, mg);
  k_qkv_gemm   <<<dim3(BT / 128, 36), 128, 0, stream>>>(xb, wn, qb, kb, vb);
  k_attention  <<<dim3(NT / 128, 4 * NH), 128, 0, stream>>>(qb, kb, vb, mg, att);
  k_outproj    <<<dim3(BT / 128, DIM / 64), 128, 0, stream>>>(att, own, out);
}
